// SpatialTranscriptomicsDecoder_3478923510056
// MI455X (gfx1250) — compile-verified, matrix operand order checked by reading, not run on hardware
//
#include <hip/hip_runtime.h>
#include <cstdint>
#include <cstddef>

// ---------------------------------------------------------------------------
// SpatialTranscriptomicsDecoder for MI455X (gfx1250, wave32, WMMA bf16)
// All GEMMs run as 16x16 tile / wave kernels on v_wmma_f32_16x16x32_bf16.
// Dominant cost is the 401MB output write of the final softplus head.
// ---------------------------------------------------------------------------

#define DEV __device__ __forceinline__

#define BB      2
#define PP      196
#define FDIM    1024
#define GG      2000
#define DD      256
#define HH      8
#define LL      4
#define DHD     32
#define HALF_D  128
#define BP      (BB*PP)     // 392
#define BG      (BB*GG)     // 4000
#define PPAD    224         // patches padded to 7 chunks of K=32

typedef __attribute__((ext_vector_type(16))) __bf16        bf16x16;
typedef __attribute__((ext_vector_type(8)))  float         f32x8;
typedef __attribute__((ext_vector_type(4)))  float         f32x4;
typedef __attribute__((ext_vector_type(4)))  unsigned int  u32x4;

union B128 { u32x4 u; __bf16 h[8]; };

DEV unsigned short f2bf(float f) {
  unsigned u = __float_as_uint(f);
  u += 0x7FFFu + ((u >> 16) & 1u);          // round-to-nearest-even
  return (unsigned short)(u >> 16);
}

// Fragment loader for the 16-bit WMMA operand layout:
// lane: idx = lane&15 (A: row M, B: col N), hl = lane>>4.
// frag[0..7]  = K = kBase + 8*hl + 0..7       (contiguous b128)
// frag[8..15] = K = kBase + 16 + 8*hl + 0..7  (contiguous b128)
DEV bf16x16 load_frag(const unsigned short* base, int lda, int row, int kBase, int hl) {
  const unsigned short* p = base + (size_t)row * lda + kBase + 8 * hl;
  B128 lo, hi;
  lo.u = *(const u32x4*)(p);
  hi.u = *(const u32x4*)(p + 16);
  bf16x16 f;
#pragma unroll
  for (int i = 0; i < 8; ++i) { f[i] = lo.h[i]; f[8 + i] = hi.h[i]; }
  return f;
}

DEV f32x8 wmma_bf16(bf16x16 a, bf16x16 b, f32x8 c) {
  return __builtin_amdgcn_wmma_f32_16x16x32_bf16(false, a, false, b, (short)0, c,
                                                 false, false);
}

// ---------------------------------------------------------------------------
// Generic GEMM: C[M,N] = A[M,K](bf16) @ W[N,K]^T(bf16) + bias
// one wave (block=32) per 16x16 output tile.
// flags: 1 = accumulate into Cf, 4 = store bf16 transposed as Vt[b,h,dh,PPAD]
// ---------------------------------------------------------------------------
__global__ void gemm16(const unsigned short* __restrict__ A,
                       const unsigned short* __restrict__ W,
                       const float* __restrict__ bias,
                       float* __restrict__ Cf,
                       unsigned short* __restrict__ Cbf,
                       int M, int N, int K, int flags) {
  const int tm = blockIdx.x, tn = blockIdx.y;
  const int lane = threadIdx.x & 31;
  const int nl = lane & 15, hl = lane >> 4;
  int rowA = tm * 16 + nl; if (rowA >= M) rowA = M - 1;
  const int rowW = tn * 16 + nl;
  f32x8 acc = {0.f,0.f,0.f,0.f,0.f,0.f,0.f,0.f};
  for (int k = 0; k < K; k += 32) {
    bf16x16 a = load_frag(A, K, rowA, k, hl);
    bf16x16 b = load_frag(W, K, rowW, k, hl);
    acc = wmma_bf16(a, b, acc);
  }
  const int n = tn * 16 + nl;
  const float bv = bias ? bias[n] : 0.0f;
#pragma unroll
  for (int r = 0; r < 8; ++r) {
    const int m = tm * 16 + r + 8 * hl;
    if (m >= M) continue;
    float v = acc[r] + bv;
    if (flags & 1) v += Cf[(size_t)m * N + n];
    if (Cf) Cf[(size_t)m * N + n] = v;
    if (Cbf) {
      if (flags & 4) {           // store V transposed for attn (contiguous-K)
        const int b_ = m / PP, p_ = m % PP;
        const int h_ = n >> 5,  dh_ = n & 31;
        Cbf[((size_t)((b_ * HH + h_) * DHD + dh_)) * PPAD + p_] = f2bf(v);
      } else {
        Cbf[(size_t)m * N + n] = f2bf(v);
      }
    }
  }
}

// ---------------------------------------------------------------------------
// Fused multi-head cross-attention: one wave per (16-gene tile, head, batch).
// scores/softmax in LDS; ctx = attn @ V via WMMA with transposed V.
// ---------------------------------------------------------------------------
__global__ void attn_fused(const unsigned short* __restrict__ Qbf,
                           const unsigned short* __restrict__ Kbf,
                           const unsigned short* __restrict__ Vt,
                           unsigned short* __restrict__ ctx) {
  __shared__ __align__(16) float          sc[16][208];
  __shared__ __align__(16) unsigned short at[16][PPAD];
  const int gt = blockIdx.x, h = blockIdx.y, b = blockIdx.z;
  const int lane = threadIdx.x & 31, nl = lane & 15, hl = lane >> 4;
  const bf16x16 qa = load_frag(Qbf, DD, b * GG + gt * 16 + nl, h * DHD, hl);
  const float scale = 0.17677669529663687f;   // 1/sqrt(32)
  for (int kt = 0; kt < 13; ++kt) {
    const int pr = kt * 16 + nl;
    const int prc = pr < PP ? pr : PP - 1;
    bf16x16 kb = load_frag(Kbf, DD, b * PP + prc, h * DHD, hl);
    f32x8 z = {0.f,0.f,0.f,0.f,0.f,0.f,0.f,0.f};
    f32x8 s = wmma_bf16(qa, kb, z);
#pragma unroll
    for (int r = 0; r < 8; ++r)
      sc[r + 8 * hl][kt * 16 + nl] = (pr < PP) ? s[r] * scale : -3.0e38f;
  }
  __syncthreads();
  if (lane < 16) {
    float mx = -3.0e38f;
    for (int j = 0; j < 208; ++j) mx = fmaxf(mx, sc[lane][j]);
    float sum = 0.f;
    for (int j = 0; j < 208; ++j) { float e = __expf(sc[lane][j] - mx); sum += e; sc[lane][j] = e; }
    const float inv = 1.f / sum;
    for (int j = 0; j < PPAD; ++j)
      at[lane][j] = (j < 208) ? f2bf(sc[lane][j] * inv) : (unsigned short)0;
  }
  __syncthreads();
  for (int nt = 0; nt < 2; ++nt) {
    f32x8 acc = {0.f,0.f,0.f,0.f,0.f,0.f,0.f,0.f};
    for (int c = 0; c < 7; ++c) {
      bf16x16 aa = load_frag(&at[0][0], PPAD, nl, c * 32, hl);
      bf16x16 vb = load_frag(Vt, PPAD, (b * HH + h) * DHD + nt * 16 + nl, c * 32, hl);
      acc = wmma_bf16(aa, vb, acc);
    }
#pragma unroll
    for (int r = 0; r < 8; ++r)
      ctx[((size_t)(b * GG + gt * 16 + r + 8 * hl)) * DD + h * DHD + nt * 16 + nl] = f2bf(acc[r]);
  }
}

// ---------------------------------------------------------------------------
// residual + LayerNorm (D=256), one wave per row; optional bf16 mirror output
// ---------------------------------------------------------------------------
__global__ void add_ln(const float* __restrict__ X, const float* __restrict__ R,
                       const float* __restrict__ gam, const float* __restrict__ bet,
                       float* __restrict__ Yf, unsigned short* __restrict__ Ybf, int M) {
  const int wave = threadIdx.x >> 5, lane = threadIdx.x & 31;
  const int row = blockIdx.x * 8 + wave;
  if (row >= M) return;
  const size_t base = (size_t)row * DD;
  float v[8]; float s = 0.f;
#pragma unroll
  for (int i = 0; i < 8; ++i) {
    const int c = lane * 8 + i;
    float x = X[base + c];
    if (R) x += R[base + c];
    v[i] = x; s += x;
  }
  for (int o = 16; o > 0; o >>= 1) s += __shfl_xor(s, o, 32);
  const float mean = s * (1.f / DD);
  float ss = 0.f;
#pragma unroll
  for (int i = 0; i < 8; ++i) { const float d = v[i] - mean; ss += d * d; }
  for (int o = 16; o > 0; o >>= 1) ss += __shfl_xor(ss, o, 32);
  const float inv = rsqrtf(ss * (1.f / DD) + 1e-5f);
#pragma unroll
  for (int i = 0; i < 8; ++i) {
    const int c = lane * 8 + i;
    const float y = (v[i] - mean) * inv * gam[c] + bet[c];
    Yf[base + c] = y;
    if (Ybf) Ybf[base + c] = f2bf(y);
  }
}

// ---------------------------------------------------------------------------
// Fused MLP + LN2: per 16-row tile (block = 256 thr = 8 waves).
// h=[16,1024] gelu'd bf16 lives in LDS; second GEMM reads it back; LN fused.
// ---------------------------------------------------------------------------
__global__ void mlp_fused(const unsigned short* __restrict__ qbf,
                          const float* __restrict__ qf,
                          const unsigned short* __restrict__ W1, const float* __restrict__ b1,
                          const unsigned short* __restrict__ W2, const float* __restrict__ b2,
                          const float* __restrict__ lng, const float* __restrict__ lnb,
                          float* __restrict__ qf_out, unsigned short* __restrict__ qbf_out) {
  __shared__ __align__(16) unsigned short Atile[16 * DD];     //  8 KB
  __shared__ __align__(16) unsigned short Htile[16 * 1024];   // 32 KB
  __shared__ __align__(16) float          Otile[16 * DD];     // 16 KB
  const int tm = blockIdx.x, tid = threadIdx.x;
  const int wave = tid >> 5, lane = tid & 31, nl = lane & 15, hl = lane >> 4;
  {
    const u32x4* src = (const u32x4*)(qbf + (size_t)tm * 16 * DD);
    u32x4* dst = (u32x4*)Atile;
    for (int i = tid; i < (16 * DD) / 8; i += 256) dst[i] = src[i];
  }
  __syncthreads();
  // stage 1: h = gelu(q @ ff1^T + b1), each wave owns 8 of 64 N-tiles
  for (int t = 0; t < 8; ++t) {
    const int nt = wave * 8 + t;
    f32x8 acc = {0.f,0.f,0.f,0.f,0.f,0.f,0.f,0.f};
    for (int kc = 0; kc < 8; ++kc) {
      bf16x16 a = load_frag(Atile, DD, nl, kc * 32, hl);
      bf16x16 b = load_frag(W1, DD, nt * 16 + nl, kc * 32, hl);
      acc = wmma_bf16(a, b, acc);
    }
    const int n = nt * 16 + nl;
    const float bv = b1[n];
#pragma unroll
    for (int r = 0; r < 8; ++r) {
      const float x = acc[r] + bv;
      const float gv = 0.5f * x * (1.f + erff(x * 0.7071067811865475f));
      Htile[(r + 8 * hl) * 1024 + n] = f2bf(gv);
    }
  }
  __syncthreads();
  // stage 2: out = h @ ff2^T + b2, each wave owns 2 of 16 N-tiles
  for (int t = 0; t < 2; ++t) {
    const int nt = wave * 2 + t;
    f32x8 acc = {0.f,0.f,0.f,0.f,0.f,0.f,0.f,0.f};
    for (int kc = 0; kc < 32; ++kc) {
      bf16x16 a = load_frag(Htile, 1024, nl, kc * 32, hl);
      bf16x16 b = load_frag(W2, 1024, nt * 16 + nl, kc * 32, hl);
      acc = wmma_bf16(a, b, acc);
    }
    const int n = nt * 16 + nl;
    const float bv = b2[n];
#pragma unroll
    for (int r = 0; r < 8; ++r) Otile[(r + 8 * hl) * DD + n] = acc[r] + bv;
  }
  __syncthreads();
  // fused residual + LN2: wave handles 2 rows
  for (int rr = 0; rr < 2; ++rr) {
    const int rloc = wave * 2 + rr;
    const size_t base = ((size_t)tm * 16 + rloc) * DD;
    float v[8]; float s = 0.f;
#pragma unroll
    for (int i = 0; i < 8; ++i) {
      const int c = lane * 8 + i;
      const float x = qf[base + c] + Otile[rloc * DD + c];
      v[i] = x; s += x;
    }
    for (int o = 16; o > 0; o >>= 1) s += __shfl_xor(s, o, 32);
    const float mean = s * (1.f / DD);
    float ss = 0.f;
#pragma unroll
    for (int i = 0; i < 8; ++i) { const float d = v[i] - mean; ss += d * d; }
    for (int o = 16; o > 0; o >>= 1) ss += __shfl_xor(ss, o, 32);
    const float inv = rsqrtf(ss * (1.f / DD) + 1e-5f);
#pragma unroll
    for (int i = 0; i < 8; ++i) {
      const int c = lane * 8 + i;
      const float y = (v[i] - mean) * inv * lng[c] + lnb[c];
      qf_out[base + c] = y;
      qbf_out[base + c] = f2bf(y);
    }
  }
}

// ---------------------------------------------------------------------------
// small helpers
// ---------------------------------------------------------------------------
__global__ void cvt_bf16(const float* __restrict__ x, unsigned short* __restrict__ y, int n) {
  const int i = blockIdx.x * 256 + threadIdx.x;
  if (i < n) y[i] = f2bf(x[i]);
}

__global__ void pe_build(const int* __restrict__ coords,
                         const float* __restrict__ rowe, const float* __restrict__ cole,
                         unsigned short* __restrict__ pebf) {
  const int i = blockIdx.x * 256 + threadIdx.x;
  if (i >= BP * DD) return;
  const int row = i / DD, c = i % DD;
  int r = coords[row * 2 + 0]; r = r < 0 ? 0 : (r > 255 ? 255 : r);
  int cc = coords[row * 2 + 1]; cc = cc < 0 ? 0 : (cc > 255 ? 255 : cc);
  const float v = (c < HALF_D) ? rowe[r * HALF_D + c] : cole[cc * HALF_D + (c - HALF_D)];
  pebf[i] = f2bf(v);
}

__global__ void q_init(const float* __restrict__ gq, float* __restrict__ q,
                       unsigned short* __restrict__ qbf) {
  const int i = blockIdx.x * 256 + threadIdx.x;
  if (i >= BG * DD) return;
  const int row = i / DD, g = row % GG, c = i % DD;
  const float v = gq[(size_t)g * DD + c];
  q[i] = v; qbf[i] = f2bf(v);
}

// ---------------------------------------------------------------------------
// Final head: out[b,p,g,:] = softplus(qh[b,g,:] + ph[b,p,:])  (401 MB write,
// bandwidth-bound; one wave per (b,p,g), 32 lanes x float4 = 128 cols)
// ---------------------------------------------------------------------------
__global__ void head_out(const float* __restrict__ qh, const float* __restrict__ ph,
                         float* __restrict__ out) {
  const int b = blockIdx.z, p = blockIdx.y;
  const int wave = threadIdx.x >> 5, lane = threadIdx.x & 31;
  const int g = blockIdx.x * 8 + wave;
  const f32x4 pv = ((const f32x4*)(ph + ((size_t)b * PP + p) * HALF_D))[lane];
  const f32x4 qv = ((const f32x4*)(qh + ((size_t)b * GG + g) * HALF_D))[lane];
  f32x4 o;
#pragma unroll
  for (int i = 0; i < 4; ++i) {
    const float x = pv[i] + qv[i];
    o[i] = x > 20.f ? x : log1pf(__expf(x));
  }
  ((f32x4*)(out + (((size_t)(b * PP + p)) * GG + g) * HALF_D))[lane] = o;
}

// ---------------------------------------------------------------------------
extern "C" void kernel_launch(void* const* d_in, const int* in_sizes, int n_in,
                              void* d_out, int out_size, void* d_ws, size_t ws_size,
                              hipStream_t stream) {
  (void)in_sizes; (void)n_in; (void)out_size; (void)ws_size;
  char* w = (char*)d_ws;
  size_t off = 0;
  auto alloc = [&](size_t bytes) -> void* {
    void* p = (void*)(w + off);
    off = (off + bytes + 255) & ~(size_t)255;
    return p;
  };
  // bf16 scratch
  unsigned short* pf_bf   = (unsigned short*)alloc((size_t)BP * FDIM * 2);
  unsigned short* wp_bf   = (unsigned short*)alloc((size_t)DD * FDIM * 2);
  unsigned short* wpe_bf  = (unsigned short*)alloc((size_t)DD * DD * 2);
  unsigned short* win_bf  = (unsigned short*)alloc((size_t)LL * 3 * DD * DD * 2);
  unsigned short* wout_bf = (unsigned short*)alloc((size_t)LL * DD * DD * 2);
  unsigned short* wff1_bf = (unsigned short*)alloc((size_t)LL * 4 * DD * DD * 2);
  unsigned short* wff2_bf = (unsigned short*)alloc((size_t)LL * DD * 4 * DD * 2);
  unsigned short* wh_bf   = (unsigned short*)alloc((size_t)HALF_D * DD * 2);
  unsigned short* pe_bf   = (unsigned short*)alloc((size_t)BP * DD * 2);
  unsigned short* pemb_bf = (unsigned short*)alloc((size_t)BP * DD * 2);
  unsigned short* q_bf    = (unsigned short*)alloc((size_t)BG * DD * 2);
  unsigned short* Q_bf    = (unsigned short*)alloc((size_t)BG * DD * 2);
  unsigned short* K_bf    = (unsigned short*)alloc((size_t)BP * DD * 2);
  unsigned short* Vt_bf   = (unsigned short*)alloc((size_t)BB * HH * DHD * PPAD * 2);
  unsigned short* ctx_bf  = (unsigned short*)alloc((size_t)BG * DD * 2);
  // f32 scratch
  float* X1  = (float*)alloc((size_t)BP * DD * 4);
  float* qf  = (float*)alloc((size_t)BG * DD * 4);
  float* att = (float*)alloc((size_t)BG * DD * 4);
  float* qh  = (float*)alloc((size_t)BG * HALF_D * 4);
  float* ph  = (float*)alloc((size_t)BP * HALF_D * 4);

  auto cvt = [&](const void* src, unsigned short* dst, int n) {
    cvt_bf16<<<(n + 255) / 256, 256, 0, stream>>>((const float*)src, dst, n);
  };
  auto gemm = [&](const unsigned short* A, const unsigned short* W,
                  const float* bias, float* Cf, unsigned short* Cbf,
                  int M, int N, int K, int flags) {
    dim3 grid((M + 15) / 16, N / 16);
    gemm16<<<grid, 32, 0, stream>>>(A, W, bias, Cf, Cbf, M, N, K, flags);
  };

  // weight + input conversions (deterministic, replayed each call)
  cvt(d_in[0],  pf_bf,   BP * FDIM);
  cvt(d_in[2],  wp_bf,   DD * FDIM);
  cvt(d_in[8],  wpe_bf,  DD * DD);
  cvt(d_in[11], win_bf,  LL * 3 * DD * DD);
  cvt(d_in[13], wout_bf, LL * DD * DD);
  cvt(d_in[15], wff1_bf, LL * 4 * DD * DD);
  cvt(d_in[17], wff2_bf, LL * DD * 4 * DD);
  cvt(d_in[23], wh_bf,   HALF_D * DD);

  // patch embedding: X1 = LN(pf @ Wp^T + b); then += pe @ Wpe^T + b
  gemm(pf_bf, wp_bf, (const float*)d_in[3], X1, nullptr, BP, DD, FDIM, 0);
  add_ln<<<(BP + 7) / 8, 256, 0, stream>>>(X1, nullptr, (const float*)d_in[4],
                                           (const float*)d_in[5], X1, nullptr, BP);
  pe_build<<<(BP * DD + 255) / 256, 256, 0, stream>>>((const int*)d_in[1],
      (const float*)d_in[6], (const float*)d_in[7], pe_bf);
  gemm(pe_bf, wpe_bf, (const float*)d_in[9], X1, pemb_bf, BP, DD, DD, 1);

  // gene queries broadcast
  q_init<<<(BG * DD + 255) / 256, 256, 0, stream>>>((const float*)d_in[10], qf, q_bf);

  for (int l = 0; l < LL; ++l) {
    const unsigned short* Wq = win_bf + (size_t)l * 3 * DD * DD;
    const unsigned short* Wk = Wq + (size_t)DD * DD;
    const unsigned short* Wv = Wk + (size_t)DD * DD;
    const float* inb = (const float*)d_in[12] + (size_t)l * 3 * DD;
    gemm(q_bf,    Wq, inb,          nullptr, Q_bf,  BG, DD, DD, 0);
    gemm(pemb_bf, Wk, inb + DD,     nullptr, K_bf,  BP, DD, DD, 0);
    gemm(pemb_bf, Wv, inb + 2 * DD, nullptr, Vt_bf, BP, DD, DD, 4);
    attn_fused<<<dim3(GG / 16, HH, BB), 32, 0, stream>>>(Q_bf, K_bf, Vt_bf, ctx_bf);
    gemm(ctx_bf, wout_bf + (size_t)l * DD * DD,
         (const float*)d_in[14] + (size_t)l * DD, att, nullptr, BG, DD, DD, 0);
    add_ln<<<(BG + 7) / 8, 256, 0, stream>>>(att, qf,
        (const float*)d_in[19] + (size_t)l * DD,
        (const float*)d_in[20] + (size_t)l * DD, qf, q_bf, BG);
    mlp_fused<<<BG / 16, 256, 0, stream>>>(q_bf, qf,
        wff1_bf + (size_t)l * 4 * DD * DD, (const float*)d_in[16] + (size_t)l * 4 * DD,
        wff2_bf + (size_t)l * DD * 4 * DD, (const float*)d_in[18] + (size_t)l * DD,
        (const float*)d_in[21] + (size_t)l * DD,
        (const float*)d_in[22] + (size_t)l * DD, qf, q_bf);
  }

  // head projections + streaming softplus output
  gemm(q_bf,    wh_bf, (const float*)d_in[24], qh, nullptr, BG, HALF_D, DD, 0);
  gemm(pemb_bf, wh_bf, nullptr,                ph, nullptr, BP, HALF_D, DD, 0);
  head_out<<<dim3(GG / 8, PP, BB), 256, 0, stream>>>(qh, ph, (float*)d_out);
}
